// OriginalMambaBlock_60361470378548
// MI455X (gfx1250) — matrix operand order repaired and checked by reading, not compiled, not run on hardware
//
#include <hip/hip_runtime.h>
#include <math.h>

// ---------------- problem constants (from reference) ----------------
#define D_MODEL   768
#define D_STATE   64
#define D_CONV    4
#define D_INNER   1536
#define HEADDIM   128
#define NHEADS    12
#define D_IN_PROJ 3212              // 2*D_INNER + 2*D_STATE + NHEADS
#define CONV_DIM  1664              // D_INNER + 2*D_STATE
#define BATCH     2
#define SEQLEN    4096
#define NTOK      (BATCH * SEQLEN)  // 8192
#define EPS_RMS   1e-5f

typedef __attribute__((ext_vector_type(16))) __bf16        v16bf;
typedef __attribute__((ext_vector_type(8)))  float         v8f;
typedef __attribute__((ext_vector_type(8)))  unsigned int  v8u;

// f32 -> bf16 round-to-nearest-even (bit trick)
__device__ __forceinline__ unsigned int f2bf(float f) {
  unsigned int u = __builtin_bit_cast(unsigned int, f);
  u += 0x7FFFu + ((u >> 16) & 1u);
  return u >> 16;
}
// load a 16-element bf16 fragment as two 16B LDS reads
__device__ __forceinline__ v16bf ld_frag(const unsigned short* rowbase,
                                         int off0, int off1) {
  uint4 lo = *(const uint4*)(rowbase + off0);
  uint4 hi = *(const uint4*)(rowbase + off1);
  v8u u;
  u[0] = lo.x; u[1] = lo.y; u[2] = lo.z; u[3] = lo.w;
  u[4] = hi.x; u[5] = hi.y; u[6] = hi.z; u[7] = hi.w;
  return __builtin_bit_cast(v16bf, u);
}
// gfx1250 async raw copy: global -> LDS, tracked by ASYNCcnt
__device__ __forceinline__ void async_cp16(unsigned lds, const void* gp) {
  asm volatile("global_load_async_to_lds_b128 %0, %1, off"
               :: "v"(lds), "v"((unsigned long long)(size_t)gp) : "memory");
}
__device__ __forceinline__ void async_cp4(unsigned lds, const void* gp) {
  asm volatile("global_load_async_to_lds_b32 %0, %1, off"
               :: "v"(lds), "v"((unsigned long long)(size_t)gp) : "memory");
}

// --------------------------------------------------------------------
// f32 -> bf16 bulk convert (n multiple of 4)
// --------------------------------------------------------------------
__global__ void cvt_bf16_kernel(const float* __restrict__ src,
                                unsigned short* __restrict__ dst, long n)
{
  long i = ((long)blockIdx.x * blockDim.x + threadIdx.x) * 4;
  if (i + 3 < n) {
    float4 v = *(const float4*)(src + i);
    uint2 r;
    r.x = f2bf(v.x) | (f2bf(v.y) << 16);
    r.y = f2bf(v.z) | (f2bf(v.w) << 16);
    *(uint2*)(dst + i) = r;
  } else {
    for (; i < n; ++i) dst[i] = (unsigned short)f2bf(src[i]);
  }
}

// --------------------------------------------------------------------
// C[M x N] = A[M x K] * B[N x K]^T; A/B bf16 K-contiguous, C f32.
// 256 threads = 8 waves; block tile 64(M) x 256(N); wave grid 2x4;
// wave tile 32x64 -> 6 fragment loads feed 8 wmma per 32-wide K step.
// LDS tiles double-buffered, filled by global_load_async_to_lds_b128
// (ASYNCcnt): tile i+1's HBM fetch overlaps tile i's WMMAs.
// One barrier + one s_wait_asynccnt per K step.
// --------------------------------------------------------------------
template<int N, int K, int LDA, int LDB, int LDC>
__global__ __launch_bounds__(256)
void gemm_bf16_wmma(const unsigned short* __restrict__ A,
                    const unsigned short* __restrict__ B,
                    float* __restrict__ C)
{
  // rows padded to 40 ushorts (80 B): 16B chunks aligned, banks spread
  __shared__ unsigned short As[2][64][40];    // 10 KB
  __shared__ unsigned short Bs[2][256][40];   // 40 KB

  const int tid  = threadIdx.x;
  const int lane = tid & 31;
  const int wave = tid >> 5;
  const int half = lane >> 4;          // 0: lanes 0-15, 1: lanes 16-31
  const int l15  = lane & 15;
  const int m_off = (wave >> 2) * 32;  // 0,32
  const int n_off = (wave & 3)  * 64;  // 0,64,128,192
  const long row0 = (long)blockIdx.x * 64;
  const long col0 = (long)blockIdx.y * 256;

  // staging coords: A 64x32 (16B/thread), B 256x32 (64B/thread, row=tid)
  const int ar = tid >> 2;             // 0..63
  const int ac = (tid & 3) * 8;        // 0,8,16,24
  const long bn   = col0 + tid;
  const long brow = (bn < (long)N) ? bn : 0;  // clamp: OOB cols never stored

  const unsigned short* aptr = A + (row0 + ar) * (long)LDA + ac;
  const unsigned short* bptr = B + brow * (long)LDB;

  const unsigned ldsA[2] = { (unsigned)(size_t)&As[0][ar][ac],
                             (unsigned)(size_t)&As[1][ar][ac] };
  const unsigned ldsB[2] = { (unsigned)(size_t)&Bs[0][tid][0],
                             (unsigned)(size_t)&Bs[1][tid][0] };

  v8f acc[2][4] = {};

  // prologue: async-fill tile 0
  async_cp16(ldsA[0], aptr);
  #pragma unroll
  for (int j = 0; j < 4; ++j) async_cp16(ldsB[0] + 16 * j, bptr + 8 * j);
  aptr += 32; bptr += 32;

  #pragma unroll 1
  for (int kk = 0; kk < K; kk += 32) {
    const int cur = (kk >> 5) & 1;
    asm volatile("s_wait_asynccnt 0" ::: "memory");
    __syncthreads();                   // tile `cur` resident for all waves

    if (kk + 32 < K) {                 // async-fill next tile into other buf
      async_cp16(ldsA[cur ^ 1], aptr);
      #pragma unroll
      for (int j = 0; j < 4; ++j) async_cp16(ldsB[cur ^ 1] + 16 * j, bptr + 8 * j);
      aptr += 32; bptr += 32;
      __builtin_prefetch(aptr + 32, 0, 1);   // hint tile i+3 toward L2
      __builtin_prefetch(bptr + 32, 0, 1);
    }

    // fragments: row/col = l15, k = 16*half + s (contiguous order, same for A and B)
    v16bf a0 = ld_frag(&As[cur][m_off +      l15][0], 16 * half, 16 * half + 8);
    v16bf a1 = ld_frag(&As[cur][m_off + 16 + l15][0], 16 * half, 16 * half + 8);
    v16bf b0 = ld_frag(&Bs[cur][n_off +      l15][0], 16 * half, 16 * half + 8);
    v16bf b1 = ld_frag(&Bs[cur][n_off + 16 + l15][0], 16 * half, 16 * half + 8);
    v16bf b2 = ld_frag(&Bs[cur][n_off + 32 + l15][0], 16 * half, 16 * half + 8);
    v16bf b3 = ld_frag(&Bs[cur][n_off + 48 + l15][0], 16 * half, 16 * half + 8);

    acc[0][0] = __builtin_amdgcn_wmma_f32_16x16x32_bf16(false, a0, false, b0,
                 (short)0, acc[0][0], false, false);
    acc[0][1] = __builtin_amdgcn_wmma_f32_16x16x32_bf16(false, a0, false, b1,
                 (short)0, acc[0][1], false, false);
    acc[0][2] = __builtin_amdgcn_wmma_f32_16x16x32_bf16(false, a0, false, b2,
                 (short)0, acc[0][2], false, false);
    acc[0][3] = __builtin_amdgcn_wmma_f32_16x16x32_bf16(false, a0, false, b3,
                 (short)0, acc[0][3], false, false);
    acc[1][0] = __builtin_amdgcn_wmma_f32_16x16x32_bf16(false, a1, false, b0,
                 (short)0, acc[1][0], false, false);
    acc[1][1] = __builtin_amdgcn_wmma_f32_16x16x32_bf16(false, a1, false, b1,
                 (short)0, acc[1][1], false, false);
    acc[1][2] = __builtin_amdgcn_wmma_f32_16x16x32_bf16(false, a1, false, b2,
                 (short)0, acc[1][2], false, false);
    acc[1][3] = __builtin_amdgcn_wmma_f32_16x16x32_bf16(false, a1, false, b3,
                 (short)0, acc[1][3], false, false);
  }

  // store D: VGPR r -> M = r + 8*half, N = l15 ; offsets compile-time
  float* cbase = C + (row0 + m_off + 8 * half) * (long)LDC + col0 + n_off + l15;
  #pragma unroll
  for (int ni = 0; ni < 4; ++ni) {
    long col = col0 + n_off + ni * 16 + l15;
    if (col < (long)N) {
      #pragma unroll
      for (int mi = 0; mi < 2; ++mi) {
        #pragma unroll
        for (int r = 0; r < 8; ++r) {
          cbase[(long)(mi * 16 + r) * LDC + ni * 16] = acc[mi][ni][r];
        }
      }
    }
  }
}

// --------------------------------------------------------------------
// depthwise causal conv (width 4) over xBC slice of zxbcdt, then SiLU
// --------------------------------------------------------------------
__global__ void conv_silu_kernel(const float* __restrict__ zx,
                                 const float* __restrict__ cw,
                                 const float* __restrict__ cb,
                                 float* __restrict__ xbc)
{
  long i = (long)blockIdx.x * blockDim.x + threadIdx.x;
  if (i >= (long)NTOK * CONV_DIM) return;
  int  c = (int)(i % CONV_DIM);
  long t = i / CONV_DIM;            // global token index
  int  l = (int)(t % SEQLEN);
  long b = t / SEQLEN;

  float acc = cb[c];
  #pragma unroll
  for (int k = 0; k < D_CONV; ++k) {
    int ls = l - (D_CONV - 1) + k;
    if (ls >= 0) {
      long row = b * SEQLEN + ls;
      acc += zx[row * (long)D_IN_PROJ + D_INNER + c] * cw[c * D_CONV + k];
    }
  }
  xbc[t * (long)CONV_DIM + c] = acc * (1.0f / (1.0f + __expf(-acc)));  // SiLU
}

// --------------------------------------------------------------------
// dt = softplus(dt_raw + dt_bias); dA = exp(dt * (-exp(A_log)))
// --------------------------------------------------------------------
__global__ void dt_kernel(const float* __restrict__ zx,
                          const float* __restrict__ dt_bias,
                          const float* __restrict__ A_log,
                          float* __restrict__ dtx, float* __restrict__ dAv)
{
  long i = (long)blockIdx.x * blockDim.x + threadIdx.x;
  if (i >= (long)NTOK * NHEADS) return;
  int  h = (int)(i % NHEADS);
  long t = i / NHEADS;
  float x  = zx[t * (long)D_IN_PROJ + (D_INNER + CONV_DIM) + h] + dt_bias[h];
  float sp = (x > 20.0f) ? x : log1pf(__expf(x));
  dtx[i] = sp;
  dAv[i] = __expf(sp * (-__expf(A_log[h])));
}

// --------------------------------------------------------------------
// sequential SSM scan: one block per (batch, head). State [128 x 64]
// in registers. x/B/C rows streamed via async global->LDS copies with
// a 4-deep LDS pipeline (issue 2 rows ahead, s_wait_asynccnt 2), so
// each row's HBM latency is covered by ~2 recurrence iterations.
// --------------------------------------------------------------------
__global__ __launch_bounds__(256)
void scan_kernel(const float* __restrict__ xbc, const float* __restrict__ dtx,
                 const float* __restrict__ dAv, const float* __restrict__ Dp,
                 float* __restrict__ y)
{
  // buf[.][0..127]=x_t, [128..191]=B_t, [192..255]=C_t
  __shared__ float buf[4][256];

  const int b   = blockIdx.x / NHEADS;
  const int h   = blockIdx.x % NHEADS;
  const int tid = threadIdx.x;
  const int p   = tid >> 1;
  const int n0  = (tid & 1) * 32;

  // per-thread source column within an xbc row
  int src;
  if (tid < HEADDIM)                src = h * HEADDIM + tid;
  else if (tid < HEADDIM + D_STATE) src = D_INNER + (tid - HEADDIM);
  else                              src = D_INNER + D_STATE + (tid - 192);

  const float* srcp = xbc + (long)b * SEQLEN * CONV_DIM + src;
  const unsigned ldsb[4] = {
    (unsigned)(size_t)&buf[0][tid], (unsigned)(size_t)&buf[1][tid],
    (unsigned)(size_t)&buf[2][tid], (unsigned)(size_t)&buf[3][tid] };

  float st[32];
  #pragma unroll
  for (int j = 0; j < 32; ++j) st[j] = 0.0f;
  const float dph = Dp[h];

  // prologue: rows 0 and 1 in flight
  async_cp4(ldsb[0], srcp);
  async_cp4(ldsb[1], srcp + (long)CONV_DIM);

  for (int t = 0; t < SEQLEN; ++t) {
    // issue row t+2 (source clamped; surplus copies land in unread bufs)
    {
      int tn = (t + 2 < SEQLEN) ? (t + 2) : (SEQLEN - 1);
      async_cp4(ldsb[(t + 2) & 3], srcp + (long)tn * CONV_DIM);
    }
    // <=2 outstanding after this: in-order completion => row t resident
    asm volatile("s_wait_asynccnt 2" ::: "memory");
    __syncthreads();

    long th = ((long)b * SEQLEN + t) * NHEADS + h;
    float dt_t = dtx[th];     // broadcast loads
    float dA_t = dAv[th];

    const float* xs = &buf[t & 3][0];
    const float* Bs = &buf[t & 3][HEADDIM];
    const float* Cs = &buf[t & 3][HEADDIM + D_STATE];

    float xv   = xs[p];
    float coef = dt_t * xv;
    float ysum = 0.0f;
    #pragma unroll
    for (int j = 0; j < 32; ++j) {
      st[j] = st[j] * dA_t + coef * Bs[n0 + j];
      ysum  = fmaf(st[j], Cs[n0 + j], ysum);
    }
    ysum += __shfl_xor(ysum, 1);
    if ((tid & 1) == 0)
      y[((long)b * SEQLEN + t) * (long)D_INNER + h * HEADDIM + p] = ysum + dph * xv;
  }
}

// --------------------------------------------------------------------
// g = (y * silu(z)) * rsqrt(mean(g^2) + eps) * norm_w ; bf16 output
// feeds the out_proj WMMA GEMM directly. One block per token.
// --------------------------------------------------------------------
__global__ __launch_bounds__(256)
void gate_norm_kernel(const float* __restrict__ y, const float* __restrict__ zx,
                      const float* __restrict__ nw,
                      unsigned short* __restrict__ g)
{
  __shared__ float red[8];
  const long t   = blockIdx.x;
  const int  tid = threadIdx.x;
  const float* yr = y  + t * (long)D_INNER;
  const float* zr = zx + t * (long)D_IN_PROJ;   // z = leading D_INNER slice

  float gv[6];
  float sq = 0.0f;
  #pragma unroll
  for (int j = 0; j < 6; ++j) {
    int i = tid + j * 256;
    float z   = zr[i];
    float val = yr[i] * (z * (1.0f / (1.0f + __expf(-z))));
    gv[j] = val;
    sq   += val * val;
  }
  #pragma unroll
  for (int s = 16; s > 0; s >>= 1) sq += __shfl_xor(sq, s);
  if ((tid & 31) == 0) red[tid >> 5] = sq;
  __syncthreads();
  float tot = 0.0f;
  #pragma unroll
  for (int w = 0; w < 8; ++w) tot += red[w];
  float scale = rsqrtf(tot / (float)D_INNER + EPS_RMS);

  unsigned short* gr = g + t * (long)D_INNER;
  #pragma unroll
  for (int j = 0; j < 6; ++j) {
    int i = tid + j * 256;
    gr[i] = (unsigned short)f2bf(gv[j] * scale * nw[i]);
  }
}

// --------------------------------------------------------------------
extern "C" void kernel_launch(void* const* d_in, const int* in_sizes, int n_in,
                              void* d_out, int out_size, void* d_ws, size_t ws_size,
                              hipStream_t stream)
{
  (void)in_sizes; (void)n_in; (void)out_size; (void)ws_size;

  const float* u       = (const float*)d_in[0];
  const float* W_in    = (const float*)d_in[1];  // [D_IN_PROJ, D_MODEL]
  const float* conv_w  = (const float*)d_in[2];  // [CONV_DIM, D_CONV]
  const float* conv_b  = (const float*)d_in[3];
  const float* dt_bias = (const float*)d_in[4];
  const float* A_log   = (const float*)d_in[5];
  const float* Dp      = (const float*)d_in[6];
  const float* norm_w  = (const float*)d_in[7];
  const float* W_out   = (const float*)d_in[8];  // [D_MODEL, D_INNER]
  float* out = (float*)d_out;

  char* ws = (char*)d_ws;
  size_t off = 0;
  float* zx  = (float*)(ws + off); off += (size_t)NTOK * D_IN_PROJ * sizeof(float);
  float* xbc = (float*)(ws + off); off += (size_t)NTOK * CONV_DIM  * sizeof(float);
  float* dtx = (float*)(ws + off); off += (size_t)NTOK * NHEADS    * sizeof(float);
  float* dAv = (float*)(ws + off); off += (size_t)NTOK * NHEADS    * sizeof(float);
  float* yv  = (float*)(ws + off); off += (size_t)NTOK * D_INNER   * sizeof(float);
  unsigned short* u_bf   = (unsigned short*)(ws + off); off += (size_t)NTOK * D_MODEL      * 2;
  unsigned short* Win_bf = (unsigned short*)(ws + off); off += (size_t)D_IN_PROJ * D_MODEL * 2;
  unsigned short* Wout_bf= (unsigned short*)(ws + off); off += (size_t)D_MODEL * D_INNER   * 2;
  unsigned short* g_bf   = (unsigned short*)xbc;  // xBC dead after scan

  const dim3 blk(256);

  // 0) one-shot bf16 conversions of GEMM operands
  {
    long n = (long)NTOK * D_MODEL;
    cvt_bf16_kernel<<<dim3((unsigned)((n / 4 + 255) / 256)), blk, 0, stream>>>(u, u_bf, n);
    n = (long)D_IN_PROJ * D_MODEL;
    cvt_bf16_kernel<<<dim3((unsigned)((n / 4 + 255) / 256)), blk, 0, stream>>>(W_in, Win_bf, n);
    n = (long)D_MODEL * D_INNER;
    cvt_bf16_kernel<<<dim3((unsigned)((n / 4 + 255) / 256)), blk, 0, stream>>>(W_out, Wout_bf, n);
  }
  // 1) in_proj: zxbcdt[8192 x 3212] = u[8192 x 768] * W_in^T
  {
    dim3 grid(NTOK / 64, (D_IN_PROJ + 255) / 256);
    gemm_bf16_wmma<D_IN_PROJ, D_MODEL, D_MODEL, D_MODEL, D_IN_PROJ>
        <<<grid, blk, 0, stream>>>(u_bf, Win_bf, zx);
  }
  // 2) depthwise causal conv + SiLU
  {
    long n = (long)NTOK * CONV_DIM;
    conv_silu_kernel<<<dim3((unsigned)((n + 255) / 256)), blk, 0, stream>>>(
        zx, conv_w, conv_b, xbc);
  }
  // 3) dt softplus / dA
  {
    long n = (long)NTOK * NHEADS;
    dt_kernel<<<dim3((unsigned)((n + 255) / 256)), blk, 0, stream>>>(
        zx, dt_bias, A_log, dtx, dAv);
  }
  // 4) sequential scan: 24 blocks (batch x head), 4-deep async pipeline
  scan_kernel<<<dim3(BATCH * NHEADS), blk, 0, stream>>>(xbc, dtx, dAv, Dp, yv);

  // 5) gate + RMSNorm -> bf16 g
  gate_norm_kernel<<<dim3(NTOK), blk, 0, stream>>>(yv, zx, norm_w, g_bf);

  // 6) out_proj: out[8192 x 768] = g[8192 x 1536] * W_out^T
  {
    dim3 grid(NTOK / 64, D_MODEL / 256);
    gemm_bf16_wmma<D_MODEL, D_INNER, D_INNER, D_INNER, D_MODEL>
        <<<grid, blk, 0, stream>>>(g_bf, Wout_bf, out);
  }
}
